// GATModel_60790967107703
// MI455X (gfx1250) — matrix operand order repaired and checked by reading, not compiled, not run on hardware
//
#include <hip/hip_runtime.h>
#include <hip/hip_bf16.h>
#include <stdint.h>
#include <limits.h>

// ---------------------------------------------------------------------------
// GAT (2-layer, 8 heads) for MI455X / gfx1250.
//  - GEMMs on v_wmma_f32_16x16x32_bf16 (fp32 accumulate); each wave owns a
//    16x64 output strip. Per K-step: load A frag + all 4 B frags into distinct
//    registers first, then issue 4 WMMAs -> staged s_wait_loadcnt instead of
//    a full wait before every WMMA.
//  - Edge softmax: int-keyed atomicMax segment-max, exp/segment-sum, normalize.
//  - Layer-1 head-mean folded into the scatter (linearity) -> no [N,2048] temp.
//  - Per-column GAT biases cancel exactly through BatchNorm -> dropped.
// ---------------------------------------------------------------------------

typedef __attribute__((ext_vector_type(16))) __bf16 v16bf;
typedef __attribute__((ext_vector_type(8)))  float  v8f;

#define HEADS 8
#define NEG_SLOPE 0.2f
#define BN_EPS 1e-5f

// ---------------- helpers ----------------

__device__ __forceinline__ uint16_t f2bf(float f) {
  uint32_t u = __float_as_uint(f);
  uint32_t r = (u + 0x7FFFu + ((u >> 16) & 1u)) >> 16;   // RNE
  return (uint16_t)r;
}

// order-preserving float -> int key (works for negatives)
__device__ __forceinline__ int fkey(float f) {
  int i = __float_as_int(f);
  return (i >= 0) ? i : (i ^ 0x7FFFFFFF);
}
__device__ __forceinline__ float fdec(int k) {
  return __int_as_float((k >= 0) ? k : (k ^ 0x7FFFFFFF));
}

__device__ __forceinline__ void edge_ep(int e, const int* ei, int Erand,
                                        int& s, int& d) {
  if (e < Erand) { s = ei[e]; d = ei[Erand + e]; }      // edge_index[0][e], [1][e]
  else { s = d = e - Erand; }                            // self loops
}

// ---------------- utility kernels ----------------

__global__ void fill_f32(float* p, float v, long n) {
  long i = (long)blockIdx.x * blockDim.x + threadIdx.x;
  if (i < n) p[i] = v;
}
__global__ void fill_i32(int* p, int v, long n) {
  long i = (long)blockIdx.x * blockDim.x + threadIdx.x;
  if (i < n) p[i] = v;
}
__global__ void cvt_f32_bf16(const float* in, uint16_t* out, long n) {
  long i = (long)blockIdx.x * blockDim.x + threadIdx.x;
  if (i < n) out[i] = f2bf(in[i]);
}
// in: [K, Ncols] row-major fp32 -> out: [Npad, K] bf16 (B^T, zero-padded rows)
__global__ void transpose_cvt_bf16(const float* in, uint16_t* out,
                                   int K, int Ncols, int Npad) {
  long i = (long)blockIdx.x * blockDim.x + threadIdx.x;
  long tot = (long)Npad * K;
  if (i >= tot) return;
  int row = (int)(i / K);          // output row = input column
  int k   = (int)(i % K);
  float v = (row < Ncols) ? in[(size_t)k * Ncols + row] : 0.f;
  out[i] = f2bf(v);
}

// ---------------- WMMA GEMM: C[M,Nvalid] = A[M,K] * Bt[Npad,K]^T (+bias) ----
// A, Bt bf16 (uint16 storage). One wave per 16x64 output strip (4 N-tiles).
// Fragment layouts per CDNA5 ISA (16-bit A 16x32 striping).

__global__ void wmma_gemm_bf16(const uint16_t* __restrict__ A,
                               const uint16_t* __restrict__ Bt,
                               float* __restrict__ C,
                               const float* __restrict__ bias,
                               int M, int K, int Nvalid, int stripsN) {
  int wave = (blockIdx.x * blockDim.x + threadIdx.x) >> 5;
  int lane = threadIdx.x & 31;
  int tileM  = wave / stripsN;
  int strip  = wave % stripsN;                 // strip of 4 N-tiles (64 cols)
  if (tileM * 16 >= M) return;                 // uniform per-wave exit

  int r  = lane & 15;
  int hf = lane >> 4;

  int arow = tileM * 16 + r; if (arow >= M) arow = M - 1;   // clamp (stores guarded)
  const __bf16* Abase = (const __bf16*)A + (size_t)arow * K;
  const __bf16* Bbase0 = (const __bf16*)Bt + (size_t)(strip * 64 + r) * K; // zero-padded

  union Frag { v16bf v; uint32_t u[8]; };
  v8f acc[4] = {{}, {}, {}, {}};

  for (int k0 = 0; k0 < K; k0 += 32) {
    // ---- load phase: A fragment + 4 B fragments into distinct registers ----
    Frag fa, fb[4];
#pragma unroll
    for (int vv = 0; vv < 8; ++vv) {
      // A (16-bit 16x32): lanes 0-15: VGPR0-3 K=0..7, VGPR4-7 K=16..23;
      // lanes 16-31 shift K by +8 within each octet.
      int ka = (vv < 4) ? (8 * hf + 2 * vv) : (16 + 8 * hf + 2 * (vv - 4));
      fa.u[vv] = *(const uint32_t*)(Abase + k0 + ka);
    }
#pragma unroll
    for (int nn = 0; nn < 4; ++nn) {
      const __bf16* Bb = Bbase0 + (size_t)(nn * 16) * K + k0;
#pragma unroll
      for (int vv = 0; vv < 8; ++vv) {
        // B (16-bit 32x16): same K striping as A (VGPR0-3 K=8*hf+0..7, VGPR4-7 K=16+8*hf+0..7).
        int kb = (vv < 4) ? (8 * hf + 2 * vv) : (16 + 8 * hf + 2 * (vv - 4));
        fb[nn].u[vv] = *(const uint32_t*)(Bb + kb);
      }
    }
    // ---- compute phase: 4 WMMAs reuse the single A fragment ----
#pragma unroll
    for (int nn = 0; nn < 4; ++nn) {
      acc[nn] = __builtin_amdgcn_wmma_f32_16x16x32_bf16(
          false, fa.v, false, fb[nn].v, (short)0, acc[nn], false, false);
    }
  }

  // C/D layout: VGPR j -> row tileM*16 + 8*hf + j, col = tile base + (lane&15)
  int rbase = tileM * 16 + 8 * hf;
#pragma unroll
  for (int nn = 0; nn < 4; ++nn) {
    int col = strip * 64 + nn * 16 + r;
    if (col >= Nvalid) continue;
    float badd = bias ? bias[col] : 0.f;
#pragma unroll
    for (int j = 0; j < 8; ++j) {
      int rr = rbase + j;
      if (rr < M) C[(size_t)rr * Nvalid + col] = acc[nn][j] + badd;
    }
  }
}

// ---------------- attention pieces ----------------

// s_src[n,h] = <h[n,h,:], a_src[h,:]>, same for dst. One thread per (n,h).
__global__ void node_scores(const float* __restrict__ h,
                            const float* __restrict__ a_src,
                            const float* __restrict__ a_dst,
                            float* __restrict__ ssrc, float* __restrict__ sdst,
                            int N, int C) {
  int idx = blockIdx.x * blockDim.x + threadIdx.x;
  if (idx >= N * HEADS) return;
  int n = idx / HEADS, hh = idx % HEADS;
  const float* hv = h + ((size_t)n * HEADS + hh) * C;
  const float* as = a_src + hh * C;
  const float* ad = a_dst + hh * C;
  float s0 = 0.f, s1 = 0.f;
  for (int c = 0; c < C; ++c) { float v = hv[c]; s0 += v * as[c]; s1 += v * ad[c]; }
  ssrc[idx] = s0; sdst[idx] = s1;
}

// pass 1: score = leaky_relu(s_src[src]+s_dst[dst]); segment max via int keys
__global__ void edge_score_max(const int* __restrict__ ei, int Erand, int Etot,
                               const float* __restrict__ ssrc,
                               const float* __restrict__ sdst,
                               float* __restrict__ score, int* __restrict__ mkey) {
  int idx = blockIdx.x * blockDim.x + threadIdx.x;
  if (idx >= Etot * HEADS) return;
  int e = idx / HEADS, h = idx % HEADS;
  int s, d; edge_ep(e, ei, Erand, s, d);
  float sc = ssrc[(size_t)s * HEADS + h] + sdst[(size_t)d * HEADS + h];
  sc = (sc > 0.f) ? sc : NEG_SLOPE * sc;
  score[idx] = sc;
  atomicMax(&mkey[(size_t)d * HEADS + h], fkey(sc));
}

// pass 2: e = exp(score - m[dst]); denom[dst] += e   (score buf -> e buf)
__global__ void edge_exp_sum(const int* __restrict__ ei, int Erand, int Etot,
                             float* __restrict__ ebuf,
                             const int* __restrict__ mkey,
                             float* __restrict__ denom) {
  int idx = blockIdx.x * blockDim.x + threadIdx.x;
  if (idx >= Etot * HEADS) return;
  int e = idx / HEADS, h = idx % HEADS;
  int s, d; edge_ep(e, ei, Erand, s, d); (void)s;
  float m = fdec(mkey[(size_t)d * HEADS + h]);
  float ev = expf(ebuf[idx] - m);
  ebuf[idx] = ev;
  atomicAdd(&denom[(size_t)d * HEADS + h], ev);
}

// pass 3 (layer0, concat): out[dst, h*C+c] += alpha * h[src, h*C+c]
// blockDim == HEADS*C (=256), one block per edge.
__global__ void edge_aggregate_concat(const int* __restrict__ ei, int Erand,
                                      const float* __restrict__ hfeat,
                                      const float* __restrict__ ebuf,
                                      const float* __restrict__ denom,
                                      float* __restrict__ out, int C) {
  __shared__ float salpha[HEADS];
  int e = blockIdx.x;
  int s, d; edge_ep(e, ei, Erand, s, d);
  int t = threadIdx.x;
  if (t < HEADS)
    salpha[t] = ebuf[(size_t)e * HEADS + t] / denom[(size_t)d * HEADS + t];
  __syncthreads();
  int h = t / C;
  float v = salpha[h] * hfeat[(size_t)s * HEADS * C + t];
  atomicAdd(&out[(size_t)d * HEADS * C + t], v);
}

// pass 3 (layer1, mean over heads folded in): out[dst,c] += (1/H) sum_h alpha_h * h[src,h,c]
// blockDim == C (=256), one block per edge.
__global__ void edge_aggregate_mean(const int* __restrict__ ei, int Erand,
                                    const float* __restrict__ hfeat,
                                    const float* __restrict__ ebuf,
                                    const float* __restrict__ denom,
                                    float* __restrict__ out, int C) {
  __shared__ float salpha[HEADS];
  int e = blockIdx.x;
  int s, d; edge_ep(e, ei, Erand, s, d);
  int t = threadIdx.x;
  if (t < HEADS)
    salpha[t] = ebuf[(size_t)e * HEADS + t] / denom[(size_t)d * HEADS + t];
  __syncthreads();
  const float* hsrc = hfeat + (size_t)s * HEADS * C;
  float acc = 0.f;
#pragma unroll
  for (int h = 0; h < HEADS; ++h) acc += salpha[h] * hsrc[h * C + t];
  atomicAdd(&out[(size_t)d * C + t], acc * (1.f / HEADS));
}

// ---------------- batch norm + elu (+residual) ----------------

// blockDim == F; each thread owns column c over a row chunk.
__global__ void bn_stats(const float* __restrict__ x, int N, int F, int rowsPerBlk,
                         float* __restrict__ sum, float* __restrict__ sumsq) {
  int c = threadIdx.x;
  int r0 = blockIdx.x * rowsPerBlk;
  int r1 = r0 + rowsPerBlk; if (r1 > N) r1 = N;
  float s = 0.f, s2 = 0.f;
  for (int r = r0; r < r1; ++r) {
    float v = x[(size_t)r * F + c];
    s += v; s2 += v * v;
  }
  atomicAdd(&sum[c], s);
  atomicAdd(&sumsq[c], s2);
}

__global__ void bn_elu_res(const float* __restrict__ x,
                           const float* __restrict__ sum,
                           const float* __restrict__ sumsq,
                           const float* __restrict__ gamma,
                           const float* __restrict__ beta,
                           const float* __restrict__ res,   // nullable
                           float* __restrict__ out, int N, int F) {
  long i = (long)blockIdx.x * blockDim.x + threadIdx.x;
  if (i >= (long)N * F) return;
  int c = (int)(i % F);
  float mu  = sum[c] / (float)N;
  float var = sumsq[c] / (float)N - mu * mu;
  float y = gamma[c] * (x[i] - mu) * rsqrtf(var + BN_EPS) + beta[c];
  y = (y > 0.f) ? y : (expf(y) - 1.f);          // ELU(alpha=1)
  if (res) y += res[i];
  out[i] = y;
}

// ---------------- host-side orchestration ----------------

static inline int cdiv(long a, long b) { return (int)((a + b - 1) / b); }

extern "C" void kernel_launch(void* const* d_in, const int* in_sizes, int n_in,
                              void* d_out, int out_size, void* d_ws, size_t ws_size,
                              hipStream_t stream) {
  const float* x      = (const float*)d_in[0];
  const int*   ei     = (const int*)  d_in[1];
  const float* W0     = (const float*)d_in[2];
  const float* a_src0 = (const float*)d_in[3];
  const float* a_dst0 = (const float*)d_in[4];
  // d_in[5] = b0: per-column bias cancels exactly through BatchNorm -> unused
  const float* g0     = (const float*)d_in[6];
  const float* be0    = (const float*)d_in[7];
  const float* W1     = (const float*)d_in[8];
  const float* a_src1 = (const float*)d_in[9];
  const float* a_dst1 = (const float*)d_in[10];
  // d_in[11] = b1: cancels through BatchNorm -> unused
  const float* g1     = (const float*)d_in[12];
  const float* be1    = (const float*)d_in[13];
  const float* Wc     = (const float*)d_in[14];
  const float* bc     = (const float*)d_in[15];
  float* logits = (float*)d_out;

  const int IN_DIM = 256, HID = 256, C0 = HID / HEADS /*32*/, C1 = HID /*256*/;
  const int NCLS = 40, NCLS_PAD = 64;            // pad to one 64-wide strip
  const int N = in_sizes[0] / IN_DIM;
  const int Erand = in_sizes[1] / 2;
  const int Etot = Erand + N;
  const int F1 = HEADS * C1;   // 2048

  // ---- bump allocator over workspace ----
  char* p = (char*)d_ws;
  auto alloc = [&](size_t bytes) -> void* {
    void* r = (void*)p;
    p += (bytes + 255) & ~(size_t)255;
    return r;
  };
  uint16_t* x_bf  = (uint16_t*)alloc((size_t)N * IN_DIM * 2); // later: hact0_bf
  uint16_t* W0t   = (uint16_t*)alloc((size_t)HID * IN_DIM * 2);
  uint16_t* W1t   = (uint16_t*)alloc((size_t)F1 * HID * 2);
  uint16_t* Wct   = (uint16_t*)alloc((size_t)NCLS_PAD * HID * 2);
  float* h0    = (float*)alloc((size_t)N * HID * 4);          // later: hfinal
  float* ssrc  = (float*)alloc((size_t)N * HEADS * 4);
  float* sdst  = (float*)alloc((size_t)N * HEADS * 4);
  int*   mkey  = (int*)  alloc((size_t)N * HEADS * 4);
  float* denom = (float*)alloc((size_t)N * HEADS * 4);
  float* ebuf  = (float*)alloc((size_t)Etot * HEADS * 4);     // later: hfinal_bf
  float* out0  = (float*)alloc((size_t)N * HID * 4);          // later: hmean
  float* hact0 = (float*)alloc((size_t)N * HID * 4);          // residual input
  float* h1    = (float*)alloc((size_t)N * F1 * 4);
  float* bnsum = (float*)alloc(HID * 4);
  float* bnsq  = (float*)alloc(HID * 4);
  (void)ws_size; (void)n_in; (void)out_size;

  const int TPB = 256;
  auto gemm = [&](const uint16_t* A, const uint16_t* Bt, float* Cp,
                  const float* bias, int M, int K, int Npad, int Nvalid) {
    int stripsN = Npad / 64;
    long waves = (long)((M + 15) / 16) * stripsN;
    wmma_gemm_bf16<<<cdiv(waves * 32, TPB), TPB, 0, stream>>>(
        A, Bt, Cp, bias, M, K, Nvalid, stripsN);
  };

  // ===================== layer 0 =====================
  cvt_f32_bf16<<<cdiv((long)N * IN_DIM, TPB), TPB, 0, stream>>>(x, x_bf, (long)N * IN_DIM);
  transpose_cvt_bf16<<<cdiv((long)HID * IN_DIM, TPB), TPB, 0, stream>>>(W0, W0t, IN_DIM, HID, HID);

  gemm(x_bf, W0t, h0, nullptr, N, IN_DIM, HID, HID);          // h0 = x @ W0  [N,256]

  node_scores<<<cdiv((long)N * HEADS, TPB), TPB, 0, stream>>>(h0, a_src0, a_dst0, ssrc, sdst, N, C0);

  fill_i32<<<cdiv((long)N * HEADS, TPB), TPB, 0, stream>>>(mkey, INT_MIN, (long)N * HEADS);
  fill_f32<<<cdiv((long)N * HEADS, TPB), TPB, 0, stream>>>(denom, 0.f, (long)N * HEADS);
  fill_f32<<<cdiv((long)N * HID, TPB), TPB, 0, stream>>>(out0, 0.f, (long)N * HID);

  edge_score_max<<<cdiv((long)Etot * HEADS, TPB), TPB, 0, stream>>>(ei, Erand, Etot, ssrc, sdst, ebuf, mkey);
  edge_exp_sum  <<<cdiv((long)Etot * HEADS, TPB), TPB, 0, stream>>>(ei, Erand, Etot, ebuf, mkey, denom);
  edge_aggregate_concat<<<Etot, HEADS * C0, 0, stream>>>(ei, Erand, h0, ebuf, denom, out0, C0);

  fill_f32<<<1, HID, 0, stream>>>(bnsum, 0.f, HID);
  fill_f32<<<1, HID, 0, stream>>>(bnsq, 0.f, HID);
  bn_stats<<<cdiv(N, 125), HID, 0, stream>>>(out0, N, HID, 125, bnsum, bnsq);
  bn_elu_res<<<cdiv((long)N * HID, TPB), TPB, 0, stream>>>(out0, bnsum, bnsq, g0, be0, nullptr, hact0, N, HID);

  // ===================== layer 1 =====================
  uint16_t* hact0_bf = x_bf;                                   // reuse
  cvt_f32_bf16<<<cdiv((long)N * HID, TPB), TPB, 0, stream>>>(hact0, hact0_bf, (long)N * HID);
  transpose_cvt_bf16<<<cdiv((long)F1 * HID, TPB), TPB, 0, stream>>>(W1, W1t, HID, F1, F1);

  gemm(hact0_bf, W1t, h1, nullptr, N, HID, F1, F1);            // h1 = h @ W1  [N,2048]

  node_scores<<<cdiv((long)N * HEADS, TPB), TPB, 0, stream>>>(h1, a_src1, a_dst1, ssrc, sdst, N, C1);

  float* hmean = out0;                                          // reuse
  fill_i32<<<cdiv((long)N * HEADS, TPB), TPB, 0, stream>>>(mkey, INT_MIN, (long)N * HEADS);
  fill_f32<<<cdiv((long)N * HEADS, TPB), TPB, 0, stream>>>(denom, 0.f, (long)N * HEADS);
  fill_f32<<<cdiv((long)N * HID, TPB), TPB, 0, stream>>>(hmean, 0.f, (long)N * HID);

  edge_score_max<<<cdiv((long)Etot * HEADS, TPB), TPB, 0, stream>>>(ei, Erand, Etot, ssrc, sdst, ebuf, mkey);
  edge_exp_sum  <<<cdiv((long)Etot * HEADS, TPB), TPB, 0, stream>>>(ei, Erand, Etot, ebuf, mkey, denom);
  edge_aggregate_mean<<<Etot, C1, 0, stream>>>(ei, Erand, h1, ebuf, denom, hmean, C1);

  float* hfinal = h0;                                           // reuse
  fill_f32<<<1, HID, 0, stream>>>(bnsum, 0.f, HID);
  fill_f32<<<1, HID, 0, stream>>>(bnsq, 0.f, HID);
  bn_stats<<<cdiv(N, 125), HID, 0, stream>>>(hmean, N, HID, 125, bnsum, bnsq);
  bn_elu_res<<<cdiv((long)N * HID, TPB), TPB, 0, stream>>>(hmean, bnsum, bnsq, g1, be1, hact0, hfinal, N, HID);

  // ===================== classifier =====================
  uint16_t* hfinal_bf = (uint16_t*)ebuf;                        // reuse (5.12MB <= 5.44MB)
  cvt_f32_bf16<<<cdiv((long)N * HID, TPB), TPB, 0, stream>>>(hfinal, hfinal_bf, (long)N * HID);
  transpose_cvt_bf16<<<cdiv((long)NCLS_PAD * HID, TPB), TPB, 0, stream>>>(Wc, Wct, HID, NCLS, NCLS_PAD);

  gemm(hfinal_bf, Wct, logits, bc, N, HID, NCLS_PAD, NCLS);     // logits [N,40] + bc
}
